// SimplePlaneDetector_18940805775921
// MI455X (gfx1250) — matrix operand order repaired and checked by reading, not compiled, not run on hardware
//
#include <hip/hip_runtime.h>

// ---------------------------------------------------------------------------
// SimplePlaneDetector for MI455X (gfx1250): bf16 WMMA implicit-GEMM convs,
// TDM-staged weight tiles (tensor_load_to_lds + s_wait_tensorcnt), LDS-staged
// activation patches, fused bias/ReLU, maxpool, 1x1 head, decode, exact
// top-k, greedy NMS.
// ---------------------------------------------------------------------------

typedef __bf16 bf16;
typedef __attribute__((ext_vector_type(16))) __bf16 v16bf;
typedef __attribute__((ext_vector_type(8)))  __bf16 v8bf;
typedef __attribute__((ext_vector_type(8)))  float  v8f;
typedef __attribute__((ext_vector_type(4)))  unsigned int u32x4;
typedef __attribute__((ext_vector_type(8)))  int i32x8;
typedef __attribute__((ext_vector_type(4)))  int i32x4;

#define NB   8      // batch
#define TOPK 200
#define SCORE_T 0.01f
#define IOU_T   0.5f

__device__ __forceinline__ bf16 f2bf(float f) {
  unsigned u = __builtin_bit_cast(unsigned, f);
  u += 0x7FFFu + ((u >> 16) & 1u);                // round-to-nearest-even
  unsigned short s = (unsigned short)(u >> 16);
  return __builtin_bit_cast(bf16, s);
}
__device__ __forceinline__ float bf2f(bf16 h) {
  unsigned short s = __builtin_bit_cast(unsigned short, h);
  unsigned u = ((unsigned)s) << 16;
  return __builtin_bit_cast(float, u);
}
__device__ __forceinline__ bf16 bf0() {
  return __builtin_bit_cast(bf16, (unsigned short)0);
}

// ---------------- elementwise f32 -> bf16 convert (images) ------------------
__global__ void cvt_f32_bf16(const float* __restrict__ src, bf16* __restrict__ dst, int n) {
  int i = blockIdx.x * blockDim.x + threadIdx.x;
  if (i < n) dst[i] = f2bf(src[i]);
}

// ------- weight repack A: OIHW f32 -> [Cout][Kpad] bf16 (layer-1 im2col) ----
__global__ void pack_weights_im2col(const float* __restrict__ w, bf16* __restrict__ wp,
                                    int K, int KPAD, int total) {
  int i = blockIdx.x * blockDim.x + threadIdx.x;
  if (i >= total) return;
  int co = i / KPAD;
  int k  = i - co * KPAD;
  wp[i] = (k < K) ? f2bf(w[(size_t)co * K + k]) : bf0();
}

// ------- weight repack B: OIHW f32 -> [tap(9)][Cout][Cin] bf16 (layers 2-4) -
__global__ void pack_weights_tap(const float* __restrict__ w, bf16* __restrict__ wp,
                                 int CIN, int COUT, int total) {
  int i = blockIdx.x * blockDim.x + threadIdx.x;
  if (i >= total) return;
  int tap = i / (COUT * CIN);
  int rem = i - tap * (COUT * CIN);
  int o = rem / CIN;
  int c = rem - o * CIN;
  wp[i] = f2bf(w[((size_t)o * CIN + c) * 9 + tap]);
}

// ---------------- layer-1 conv (CIN=3) via im2col WMMA ----------------------
template<int CIN, int COUT, int HI, int WI, int NW>
__global__ __launch_bounds__(NW * 32)
void conv3x3_im2col_wmma(const bf16* __restrict__ X, const bf16* __restrict__ Wp,
                         const float* __restrict__ bias, bf16* __restrict__ Y)
{
  constexpr int K    = CIN * 9;
  constexpr int KPAD = (K + 31) & ~31;
  constexpr int CT   = COUT / 16;

  __shared__ alignas(64) bf16 Bs[16][32];

  const int wave = threadIdx.x >> 5;
  const int lane = threadIdx.x & 31;
  const int n    = lane & 15;
  const int hh   = lane >> 4;

  const int w0 = blockIdx.x * 16;
  const int h  = blockIdx.y;
  const int zb = blockIdx.z;
  const int b     = zb / (CT / NW);
  const int cout0 = ((zb % (CT / NW)) * NW + wave) * 16;

  const bf16* Wrow = Wp + (size_t)(cout0 + n) * KPAD;

  v8f acc = {0.f, 0.f, 0.f, 0.f, 0.f, 0.f, 0.f, 0.f};

  for (int kk = 0; kk < KPAD; kk += 32) {
    for (int e = threadIdx.x; e < 512; e += NW * 32) {
      const int kloc = e >> 4;
      const int nn   = e & 15;
      const int k    = kk + kloc;
      bf16 v = bf0();
      if (k < K) {
        const int c  = k / 9;
        const int rs = k - c * 9;
        const int r  = rs / 3;
        const int s  = rs - r * 3;
        const int hy = h + r - 1;
        const int wx = w0 + nn + s - 1;
        if (hy >= 0 && hy < HI && wx >= 0 && wx < WI)
          v = X[((size_t)b * CIN + c) * (size_t)(HI * WI) + (size_t)hy * WI + wx];
      }
      Bs[nn][kloc] = v;
    }
    __syncthreads();

    const v16bf afrag = *(const v16bf*)(Wrow + kk + hh * 16);
    const v16bf bfrag = *(const v16bf*)&Bs[n][hh * 16];

    acc = __builtin_amdgcn_wmma_f32_16x16x32_bf16(false, afrag, false, bfrag,
                                                  (short)0, acc, false, false);
    __syncthreads();
  }

#pragma unroll
  for (int v = 0; v < 8; ++v) {
    const int mm = v + 8 * hh;
    float val = fmaxf(acc[v] + bias[cout0 + mm], 0.f);
    Y[(((size_t)b * COUT + cout0 + mm) * HI + h) * WI + w0 + n] = f2bf(val);
  }
}

// ---------------- layers 2-4: tap-decomposed WMMA conv ----------------------
// Each wave: 16 cout x 64 positions (4 accumulators). Per 32-channel chunk:
//  - wave issues a TDM tensor_load_to_lds of its [9][16][32] weight tile
//  - block cooperatively stages transposed activation patch [3][66][32ch]
//  - 9 taps x 4 subtiles = 36 WMMAs per wave per chunk
template<int CIN, int COUT, int HI, int WI, int NW>
__global__ __launch_bounds__(NW * 32)
void conv3x3_tap_wmma(const bf16* __restrict__ X, const bf16* __restrict__ Wt,
                      const float* __restrict__ bias, bf16* __restrict__ Y)
{
  constexpr int CT   = COUT / 16;
  constexpr int COLS = 66;                       // w0-1 .. w0+64 halo incl.

  __shared__ alignas(64) bf16 Pa[3][COLS][32];   // [row][col][chan] transposed
  __shared__ alignas(64) bf16 Wl[NW][9][16][32]; // per-wave [tap][cout][cin]

  const int wave = threadIdx.x >> 5;
  const int lane = threadIdx.x & 31;
  const int nn   = lane & 15;
  const int hh   = lane >> 4;

  const int w0 = blockIdx.x * 64;
  const int h  = blockIdx.y;
  const int zb = blockIdx.z;
  const int b     = zb / (CT / NW);
  const int cout0 = ((zb % (CT / NW)) * NW + wave) * 16;

  v8f acc[4];
#pragma unroll
  for (int t = 0; t < 4; ++t)
    acc[t] = (v8f){0.f, 0.f, 0.f, 0.f, 0.f, 0.f, 0.f, 0.f};

  for (int cc = 0; cc < CIN; cc += 32) {
    // ---- 1) TDM: async DMA this wave's weight tile into LDS ----
    {
      unsigned long long ga =
          (unsigned long long)(const void*)(Wt + (size_t)cout0 * CIN + cc);
      unsigned lds_base = (unsigned)(unsigned long long)(void*)&Wl[wave][0][0][0];
      unsigned long long td1s = (unsigned long long)COUT * CIN; // tap stride
      u32x4 g0;
      g0[0] = 1u;                                   // count=1, user descriptor
      g0[1] = lds_base;                             // lds_addr
      g0[2] = (unsigned)ga;                         // global_addr[31:0]
      g0[3] = (unsigned)((ga >> 32) & 0x01FFFFFFull) | 0x80000000u; // type=2
      i32x8 g1;
      g1[0] = (int)(1u << 16);                      // data_size = 2 bytes
      g1[1] = (int)(((unsigned)CIN & 0xFFFFu) << 16);          // tensor_dim0 lo
      g1[2] = (int)(((unsigned)CIN >> 16) | (((unsigned)COUT & 0xFFFFu) << 16));
      g1[3] = (int)(((unsigned)COUT >> 16) | (32u << 16));     // tile_dim0=32
      g1[4] = (int)(16u | (9u << 16));              // tile_dim1=16, tile_dim2=9
      g1[5] = (int)(unsigned)CIN;                   // dim0 stride (cout step)
      g1[6] = (int)((unsigned)(td1s & 0xFFFFull) << 16);       // dim1 stride lo
      g1[7] = (int)(unsigned)(td1s >> 16);                     // dim1 stride hi
      i32x4 g2;
      g2[0] = 9; g2[1] = 0; g2[2] = 0; g2[3] = 0;   // tensor_dim2=9
      i32x4 g3;
      g3[0] = 0; g3[1] = 0; g3[2] = 0; g3[3] = 0;
      i32x8 g4;                                     // 6-arg toolchain: extra
      g4[0] = 0; g4[1] = 0; g4[2] = 0; g4[3] = 0;   // group, zero-filled
      g4[4] = 0; g4[5] = 0; g4[6] = 0; g4[7] = 0;
      __builtin_amdgcn_tensor_load_to_lds(g0, g1, g2, g3, g4, 0);
    }

    // ---- 2) cooperative transposed staging of the activation patch ----
    for (int p = threadIdx.x; p < 96; p += NW * 32) {
      const int c = p & 31;
      const int r = p >> 5;
      const int hy = h + r - 1;
      if (hy < 0 || hy >= HI) {
#pragma unroll 6
        for (int col = 0; col < COLS; ++col) Pa[r][col][c] = bf0();
      } else {
        const bf16* row = X + ((size_t)b * CIN + cc + c) * (size_t)(HI * WI)
                            + (size_t)hy * WI + w0;
        Pa[r][0][c] = (w0 > 0) ? row[-1] : bf0();
#pragma unroll
        for (int v = 0; v < 8; ++v) {
          const v8bf d = *(const v8bf*)(row + v * 8);
#pragma unroll
          for (int j = 0; j < 8; ++j) Pa[r][1 + v * 8 + j][c] = d[j];
        }
        Pa[r][65][c] = (w0 + 64 < WI) ? row[64] : bf0();
        if (cc + 32 < CIN)                          // prefetch next chunk row
          __builtin_prefetch(row + 32 * (size_t)(HI * WI), 0, 1);
      }
    }
    __syncthreads();

    // ---- 3) wait for TDM weight tile ----
    __builtin_amdgcn_s_wait_tensorcnt(0);
    asm volatile("" ::: "memory");

    // ---- 4) 9 taps x 4 N-subtiles of WMMA ----
#pragma unroll
    for (int tap = 0; tap < 9; ++tap) {
      const int r = tap / 3;
      const int s = tap - r * 3;
      const bf16* wrow = &Wl[wave][tap][nn][0];     // A row M = lane&15
      const v16bf afrag = *(const v16bf*)(wrow + hh * 16);
#pragma unroll
      for (int t = 0; t < 4; ++t) {
        const v16bf bfrag = *(const v16bf*)&Pa[r][t * 16 + nn + s][hh * 16];
        acc[t] = __builtin_amdgcn_wmma_f32_16x16x32_bf16(false, afrag, false, bfrag,
                                                         (short)0, acc[t], false, false);
      }
    }
    __syncthreads();
  }

  // ---- epilogue: bias + ReLU, bf16 store (D: lane n = N, vgpr v -> M) ----
#pragma unroll
  for (int t = 0; t < 4; ++t) {
#pragma unroll
    for (int v = 0; v < 8; ++v) {
      const int mm = v + 8 * hh;
      float val = fmaxf(acc[t][v] + bias[cout0 + mm], 0.f);
      Y[(((size_t)b * COUT + cout0 + mm) * HI + h) * WI + w0 + t * 16 + nn] = f2bf(val);
    }
  }
}

// ---------------- 2x2 max pool, stride 2 ------------------------------------
__global__ void maxpool2x2(const bf16* __restrict__ Y, bf16* __restrict__ P,
                           int C, int HI, int WI, int total) {
  int i = blockIdx.x * blockDim.x + threadIdx.x;
  if (i >= total) return;
  const int WO = WI >> 1, HO = HI >> 1;
  int x = i % WO;
  int t = i / WO;
  int y = t % HO; t /= HO;
  int c = t % C;
  int b = t / C;
  const bf16* src = Y + (((size_t)b * C + c) * HI + 2 * y) * WI + 2 * x;
  float v0 = bf2f(src[0]),  v1 = bf2f(src[1]);
  float v2 = bf2f(src[WI]), v3 = bf2f(src[WI + 1]);
  P[i] = f2bf(fmaxf(fmaxf(v0, v1), fmaxf(v2, v3)));
}

// ---------------- 1x1 head conv (256 -> 5), f32 out -------------------------
__global__ void head1x1(const bf16* __restrict__ X, const float* __restrict__ wh,
                        const float* __restrict__ bh, float* __restrict__ pred) {
  int idx = blockIdx.x * blockDim.x + threadIdx.x;   // NB*5*1024
  if (idx >= NB * 5 * 1024) return;
  int cell = idx & 1023;
  int bo   = idx >> 10;
  int o = bo % 5, b = bo / 5;
  float acc = bh[o];
  const bf16* xp = X + (size_t)b * 256 * 1024 + cell;
  const float* wr = wh + o * 256;
#pragma unroll 8
  for (int c = 0; c < 256; ++c)
    acc += bf2f(xp[(size_t)c * 1024]) * wr[c];
  pred[((size_t)b * 5 + o) * 1024 + cell] = acc;
}

// ---------------- decode: boxes + masked scores -----------------------------
__global__ void decode(const float* __restrict__ pred, float* __restrict__ boxes,
                       float* __restrict__ msc) {
  int idx = blockIdx.x * blockDim.x + threadIdx.x;   // NB*1024
  if (idx >= NB * 1024) return;
  int cell = idx & 1023;
  int b    = idx >> 10;
  const float* pb = pred + (size_t)b * 5 * 1024 + cell;
  float p0 = pb[0], p1 = pb[1024], p2 = pb[2 * 1024], p3 = pb[3 * 1024], p4 = pb[4 * 1024];
  float obj = 1.f / (1.f + expf(-p0));
  float tx  = 1.f / (1.f + expf(-p1));
  float ty  = 1.f / (1.f + expf(-p2));
  float bw  = expf(p3) * 16.f;
  float bh  = expf(p4) * 16.f;
  float gx = (float)(cell & 31), gy = (float)(cell >> 5);
  float cx = gx * 16.f + tx * 16.f;
  float cy = gy * 16.f + ty * 16.f;
  float x1 = fminf(fmaxf(cx - bw * 0.5f, 0.f), 511.f);
  float y1 = fminf(fmaxf(cy - bh * 0.5f, 0.f), 511.f);
  float x2 = fminf(fmaxf(cx + bw * 0.5f, 0.f), 511.f);
  float y2 = fminf(fmaxf(cy + bh * 0.5f, 0.f), 511.f);
  float* bx = boxes + (size_t)idx * 4;
  bx[0] = x1; bx[1] = y1; bx[2] = x2; bx[3] = y2;
  msc[idx] = (obj >= SCORE_T) ? obj : -1.f;
}

// ---------------- exact top-200 via rank-by-count (matches lax.top_k) -------
__global__ __launch_bounds__(1024)
void topk200(const float* __restrict__ boxes, const float* __restrict__ msc,
             float* __restrict__ boxesK, float* __restrict__ svals) {
  __shared__ float s[1024];
  const int b = blockIdx.x;
  const int i = threadIdx.x;
  s[i] = msc[b * 1024 + i];
  __syncthreads();
  const float si = s[i];
  int rank = 0;
  for (int j = 0; j < 1024; ++j) {
    float sj = s[j];
    rank += (sj > si) || (sj == si && j < i);
  }
  if (rank < TOPK) {
    svals[b * TOPK + rank] = si;
    const float* src = boxes + ((size_t)b * 1024 + i) * 4;
    float* dst = boxesK + ((size_t)b * TOPK + rank) * 4;
    dst[0] = src[0]; dst[1] = src[1]; dst[2] = src[2]; dst[3] = src[3];
  }
}

// ---------------- greedy NMS + final output ---------------------------------
__global__ __launch_bounds__(256)
void nms_finalize(const float* __restrict__ boxesK, const float* __restrict__ svals,
                  float* __restrict__ out) {
  __shared__ float x1[TOPK], y1[TOPK], x2[TOPK], y2[TOPK], ar[TOPK], sc[TOPK];
  __shared__ int keep[TOPK];
  const int b = blockIdx.x;
  const int t = threadIdx.x;
  if (t < TOPK) {
    const float* src = boxesK + ((size_t)b * TOPK + t) * 4;
    x1[t] = src[0]; y1[t] = src[1]; x2[t] = src[2]; y2[t] = src[3];
    sc[t] = svals[b * TOPK + t];
    ar[t] = (x2[t] - x1[t]) * (y2[t] - y1[t]);
    keep[t] = (sc[t] >= SCORE_T) ? 1 : 0;
  }
  for (int i = 0; i < TOPK; ++i) {
    __syncthreads();
    if (t < TOPK && t > i && keep[i]) {
      float xx1 = fmaxf(x1[i], x1[t]);
      float yy1 = fmaxf(y1[i], y1[t]);
      float xx2 = fminf(x2[i], x2[t]);
      float yy2 = fminf(y2[i], y2[t]);
      float inter = fmaxf(xx2 - xx1, 0.f) * fmaxf(yy2 - yy1, 0.f);
      float uni   = ar[i] + ar[t] - inter;
      float iou   = inter / fmaxf(uni, 1e-6f);
      if (iou > IOU_T) keep[t] = 0;
    }
  }
  __syncthreads();
  if (t < TOPK) {
    float kf = keep[t] ? 1.f : 0.f;
    float sv = (sc[t] >= SCORE_T) ? sc[t] : 0.f;
    float* o5 = out + ((size_t)b * TOPK + t) * 5;
    o5[0] = x1[t] * kf; o5[1] = y1[t] * kf; o5[2] = x2[t] * kf; o5[3] = y2[t] * kf;
    o5[4] = sv * kf;
    out[(size_t)NB * TOPK * 5 + b * TOPK + t] = kf;   // keep flags as floats
  }
}

// ---------------------------------------------------------------------------
extern "C" void kernel_launch(void* const* d_in, const int* in_sizes, int n_in,
                              void* d_out, int out_size, void* d_ws, size_t ws_size,
                              hipStream_t stream) {
  const float* images = (const float*)d_in[0];
  const float* w1 = (const float*)d_in[1];  const float* b1 = (const float*)d_in[2];
  const float* w2 = (const float*)d_in[3];  const float* b2 = (const float*)d_in[4];
  const float* w3 = (const float*)d_in[5];  const float* b3 = (const float*)d_in[6];
  const float* w4 = (const float*)d_in[7];  const float* b4 = (const float*)d_in[8];
  const float* wh = (const float*)d_in[9];  const float* bh = (const float*)d_in[10];

  char* ws = (char*)d_ws;
  auto alloc = [&](size_t bytes) -> char* {
    char* p = ws;
    ws += (bytes + 255) & ~(size_t)255;
    return p;
  };

  bf16* x0    = (bf16*)alloc((size_t)NB * 3 * 512 * 512 * 2);      // 12.6 MB
  bf16* yfull = (bf16*)alloc((size_t)NB * 32 * 512 * 512 * 2);     // 134 MB (reused)
  bf16* ping  = (bf16*)alloc((size_t)NB * 32 * 256 * 256 * 2);     // 33.6 MB (p1, p3)
  bf16* pong  = (bf16*)alloc((size_t)NB * 64 * 128 * 128 * 2);     // 16.8 MB (p2, p4)
  bf16* w1p   = (bf16*)alloc((size_t)32  * 32   * 2);              // [Cout][Kpad]
  bf16* w2p   = (bf16*)alloc((size_t)9 * 64  * 32  * 2);           // [9][Cout][Cin]
  bf16* w3p   = (bf16*)alloc((size_t)9 * 128 * 64  * 2);
  bf16* w4p   = (bf16*)alloc((size_t)9 * 256 * 128 * 2);
  float* pred   = (float*)alloc((size_t)NB * 5 * 1024 * 4);
  float* boxes  = (float*)alloc((size_t)NB * 1024 * 4 * 4);
  float* msc    = (float*)alloc((size_t)NB * 1024 * 4);
  float* boxesK = (float*)alloc((size_t)NB * TOPK * 4 * 4);
  float* svals  = (float*)alloc((size_t)NB * TOPK * 4);

  // --- convert inputs to bf16 / pack weights ---
  {
    int n = NB * 3 * 512 * 512;
    cvt_f32_bf16<<<(n + 255) / 256, 256, 0, stream>>>(images, x0, n);
  }
  pack_weights_im2col<<<((32 * 32) + 255) / 256, 256, 0, stream>>>(w1, w1p, 27, 32, 32 * 32);
  pack_weights_tap<<<((9 * 64  * 32)  + 255) / 256, 256, 0, stream>>>(w2, w2p, 32,  64,  9 * 64  * 32);
  pack_weights_tap<<<((9 * 128 * 64)  + 255) / 256, 256, 0, stream>>>(w3, w3p, 64,  128, 9 * 128 * 64);
  pack_weights_tap<<<((9 * 256 * 128) + 255) / 256, 256, 0, stream>>>(w4, w4p, 128, 256, 9 * 256 * 128);

  // --- layer 1: 3 -> 32, 512x512, pool -> 256x256 ---
  conv3x3_im2col_wmma<3, 32, 512, 512, 2>
      <<<dim3(512 / 16, 512, NB * (32 / 16 / 2)), 2 * 32, 0, stream>>>(x0, w1p, b1, yfull);
  {
    int n = NB * 32 * 256 * 256;
    maxpool2x2<<<(n + 255) / 256, 256, 0, stream>>>(yfull, ping, 32, 512, 512, n);
  }
  // --- layer 2: 32 -> 64, 256x256, pool -> 128x128 ---
  conv3x3_tap_wmma<32, 64, 256, 256, 4>
      <<<dim3(256 / 64, 256, NB * (64 / 16 / 4)), 4 * 32, 0, stream>>>(ping, w2p, b2, yfull);
  {
    int n = NB * 64 * 128 * 128;
    maxpool2x2<<<(n + 255) / 256, 256, 0, stream>>>(yfull, pong, 64, 256, 256, n);
  }
  // --- layer 3: 64 -> 128, 128x128, pool -> 64x64 ---
  conv3x3_tap_wmma<64, 128, 128, 128, 4>
      <<<dim3(128 / 64, 128, NB * (128 / 16 / 4)), 4 * 32, 0, stream>>>(pong, w3p, b3, yfull);
  {
    int n = NB * 128 * 64 * 64;
    maxpool2x2<<<(n + 255) / 256, 256, 0, stream>>>(yfull, ping, 128, 128, 128, n);
  }
  // --- layer 4: 128 -> 256, 64x64, pool -> 32x32 ---
  conv3x3_tap_wmma<128, 256, 64, 64, 4>
      <<<dim3(64 / 64, 64, NB * (256 / 16 / 4)), 4 * 32, 0, stream>>>(ping, w4p, b4, yfull);
  {
    int n = NB * 256 * 32 * 32;
    maxpool2x2<<<(n + 255) / 256, 256, 0, stream>>>(yfull, pong, 256, 64, 64, n);
  }

  // --- head + decode + topk + nms ---
  {
    int n = NB * 5 * 1024;
    head1x1<<<(n + 255) / 256, 256, 0, stream>>>(pong, wh, bh, pred);
  }
  {
    int n = NB * 1024;
    decode<<<(n + 255) / 256, 256, 0, stream>>>(pred, boxes, msc);
  }
  topk200<<<NB, 1024, 0, stream>>>(boxes, msc, boxesK, svals);
  nms_finalize<<<NB, 256, 0, stream>>>(boxesK, svals, (float*)d_out);
}
